// StackedLSTM_43215960933289
// MI455X (gfx1250) — matrix operand order repaired and checked by reading, not compiled, not run on hardware
//
#include <hip/hip_runtime.h>

// ---------------------------------------------------------------------------
// Stacked LSTM (L=2, B=64, T=256, H=1024) for gfx1250 (MI455X, wave32, WMMA)
//   * bf16 operands (weights packed to WMMA fragment order; activations
//     pre-converted), fp32 accumulation via v_wmma_f32_16x16x32_bf16
//   * B (weight) fragments staged through LDS: one global read per workgroup
//     instead of one per mw-wave (4x less L2 weight traffic per step)
//   * software-pipelined: next round's B tiles prefetched to registers while
//     WMMAs consume current LDS buffer (double-buffered, 1 barrier/round)
//   * fused 4-gate epilogue: activations + c/h update in registers
//   * recurrence = 512 dependent launches on `stream` (graph-capture safe)
// ---------------------------------------------------------------------------

typedef __attribute__((ext_vector_type(16))) __bf16 v16bf;
typedef __attribute__((ext_vector_type(8)))  __bf16 v8bf;
typedef __attribute__((ext_vector_type(8)))  float  v8f;

#define LSTM_B 64
#define LSTM_T 256
#define LSTM_H 1024
#define LSTM_G 4096 /* 4*H */

// ---------------------------------------------------------------------------
// Pack W[K=1024][N=4096] (f32 row-major) into bf16 WMMA B-fragment tiles:
//   idx = (((kc*256 + nc)*32 + lane)*16 + e)
//   lane <  16: N = nc*16 + lane,      K = kc*32 + e
//   lane >= 16: N = nc*16 + lane - 16, K = kc*32 + 16 + e
// ---------------------------------------------------------------------------
__global__ __launch_bounds__(256) void pack_w_bf16(const float* __restrict__ W,
                                                   __bf16* __restrict__ P) {
  long idx = (long)blockIdx.x * blockDim.x + threadIdx.x;   // 2^22 total
  int e    = (int)(idx & 15);
  int lane = (int)((idx >> 4) & 31);
  int nc   = (int)((idx >> 9) & 255);
  int kc   = (int)(idx >> 17);
  int n = nc * 16 + (lane & 15);
  int k = kc * 32 + ((lane >> 4) ? 16 : 0) + e;
  P[idx] = (__bf16)W[(long)k * LSTM_G + n];
}

__global__ __launch_bounds__(256) void to_bf16(const float* __restrict__ src,
                                               __bf16* __restrict__ dst, long n) {
  long i = (long)blockIdx.x * blockDim.x + threadIdx.x;
  if (i < n) dst[i] = (__bf16)src[i];
}

__global__ __launch_bounds__(256) void bias_sum(const float* __restrict__ bih,
                                                const float* __restrict__ bhh,
                                                float* __restrict__ bsum, int n) {
  int i = blockIdx.x * blockDim.x + threadIdx.x;
  if (i < n) bsum[i] = bih[i] + bhh[i];
}

__device__ __forceinline__ float sigmoid_(float x) {
  return 1.0f / (1.0f + __expf(-x));
}

// ---------------------------------------------------------------------------
// One LSTM cell step for one layer.
//   grid = 32 blocks x 256 threads (8 waves). Block b covers hidden cols
//   [32b, 32b+32) across all 4 gates. Wave: mw = wave&3 (16 batch rows),
//   nw = wave>>2 (16-col subtile). 4 accumulators per wave (i,f,g,o).
//
//   LDS staging: 8 B-tiles per K-chunk (tile = g*2 + ncsub), 4 K-chunks per
//   round, double buffered: 2*4*8*32*16 bf16 = 64 KB. Cooperative prefetch:
//   thread(tid) handles (kcl=it, tile=wave, lane=lane) -> 4x 32B per round.
// ---------------------------------------------------------------------------
__global__ __launch_bounds__(256) void lstm_step(
    const __bf16* __restrict__ in, long in_rs, long in_off,
    const __bf16* __restrict__ hprev,
    const __bf16* __restrict__ wihP, const __bf16* __restrict__ whhP,
    const float* __restrict__ bsum,
    float* __restrict__ cbuf, float* __restrict__ hfull,
    __bf16* __restrict__ hnext_bf,
    float* __restrict__ out, long out_rs, long out_off) {
  __shared__ __bf16 smem[2 * 4 * 8 * 32 * 16];   // [par][kcl][tile][lane][e]

  const int lane    = threadIdx.x & 31;
  const int wave    = threadIdx.x >> 5;
  const int mw      = wave & 3;
  const int nw      = wave >> 2;
  const int l15     = lane & 15;
  const int halfsel = lane >> 4;
  const int colbase = blockIdx.x * 32 + nw * 16;   // hidden col base in [0,1024)
  const int arow    = mw * 16 + l15;               // this lane's A-matrix row

  // Cooperative-prefetch tile index for this thread (tile == wave):
  //   nc_s = (wave>>1)*64 + blockIdx.x*2 + (wave&1)
  const int nc_s = (wave >> 1) * 64 + blockIdx.x * 2 + (wave & 1);

  v8f acc[4] = {v8f{0.f,0.f,0.f,0.f,0.f,0.f,0.f,0.f},
                v8f{0.f,0.f,0.f,0.f,0.f,0.f,0.f,0.f},
                v8f{0.f,0.f,0.f,0.f,0.f,0.f,0.f,0.f},
                v8f{0.f,0.f,0.f,0.f,0.f,0.f,0.f,0.f}};

  // Two GEMM phases accumulate into the same tiles: x@Wih then h@Whh.
  for (int phase = 0; phase < 2; ++phase) {
    const __bf16* A    = phase ? hprev        : in;
    const long    a_rs = phase ? (long)LSTM_H : in_rs;
    const long    a_of = phase ? 0L           : in_off;
    const __bf16* W    = phase ? whhP         : wihP;
    // lane<16 -> K runs {kc*32+0..15}; lane>=16 -> {+16..31}
    const __bf16* aprow = A + (long)arow * a_rs + a_of + (halfsel ? 16 : 0);

    // ---- prologue: prefetch round 0 (kc = 0..3) ----
    v16bf pre[4];
#pragma unroll
    for (int it = 0; it < 4; ++it)
      pre[it] = *(const v16bf*)(W + (((long)it * 256 + nc_s) * 32 + lane) * 16);

    for (int r = 0; r < 8; ++r) {              // 8 rounds x 4 kc = K/32
      const int par = r & 1;
      // store prefetched B tiles to LDS
#pragma unroll
      for (int it = 0; it < 4; ++it)
        *(v16bf*)&smem[(((par * 4 + it) * 8 + wave) * 32 + lane) * 16] = pre[it];
      // prefetch next round's B tiles (overlaps with consume below)
      if (r < 7) {
#pragma unroll
        for (int it = 0; it < 4; ++it) {
          const long kc = (long)(r + 1) * 4 + it;
          pre[it] = *(const v16bf*)(W + ((kc * 256 + nc_s) * 32 + lane) * 16);
        }
      }
      __syncthreads();                         // stores visible to all waves

      // consume: 4 K-chunks, 4 gates each
#pragma unroll
      for (int kcl = 0; kcl < 4; ++kcl) {
        const int kc = r * 4 + kcl;
        const __bf16* ap = aprow + kc * 32;
        v8bf r0 = *(const v8bf*)(ap);          // 16B: K run 0
        v8bf r1 = *(const v8bf*)(ap + 8);      // 16B: K run 1
        v16bf afrag = __builtin_shufflevector(
            r0, r1, 0, 1, 2, 3, 4, 5, 6, 7, 8, 9, 10, 11, 12, 13, 14, 15);
#pragma unroll
        for (int g = 0; g < 4; ++g) {
          const int tile = g * 2 + nw;
          const __bf16* sp =
              &smem[(((par * 4 + kcl) * 8 + tile) * 32 + lane) * 16];
          v16bf bfrag = *(const v16bf*)sp;     // 2x ds_load_b128 (broadcast)
          acc[g] = __builtin_amdgcn_wmma_f32_16x16x32_bf16(
              /*neg_a=*/false, afrag, /*neg_b=*/false, bfrag,
              /*c_mod=*/(short)0, acc[g], /*reuse_a=*/false, /*reuse_b=*/false);
        }
      }
    }
    __syncthreads();                           // phase boundary safety
  }

  // --- Fused epilogue: biases, activations, c/h update (C/D f32 layout) ---
#pragma unroll
  for (int j = 0; j < 8; ++j) {
    const int row = mw * 16 + j + (halfsel ? 8 : 0);   // batch index
    const int col = colbase + l15;                     // hidden index
    const float gi = acc[0][j] + bsum[0 * LSTM_H + col];
    const float gf = acc[1][j] + bsum[1 * LSTM_H + col];
    const float gg = acc[2][j] + bsum[2 * LSTM_H + col];
    const float go = acc[3][j] + bsum[3 * LSTM_H + col];
    const long  hc = (long)row * LSTM_H + col;
    const float cprev = cbuf[hc];
    const float cn = sigmoid_(gf) * cprev + sigmoid_(gi) * tanhf(gg);
    const float hn = sigmoid_(go) * tanhf(cn);
    cbuf[hc]     = cn;                 // in-place (same tile reads/writes)
    hfull[hc]    = hn;                 // fp32 copy (final state)
    hnext_bf[hc] = (__bf16)hn;         // feeds next step + next layer GEMMs
    if (out) out[(long)row * out_rs + out_off + col] = hn;
  }
}

// ---------------------------------------------------------------------------
// Host-side orchestration (graph-capture safe: kernels + hipMemsetAsync +
// hipMemcpyAsync on `stream` only; deterministic every call).
// ---------------------------------------------------------------------------
extern "C" void kernel_launch(void* const* d_in, const int* in_sizes, int n_in,
                              void* d_out, int out_size, void* d_ws, size_t ws_size,
                              hipStream_t stream) {
  (void)in_sizes; (void)n_in; (void)out_size; (void)ws_size;
  const float* x   = (const float*)d_in[0];   // [B, T, H]
  const float* Wih = (const float*)d_in[1];   // [L, H, 4H]
  const float* Whh = (const float*)d_in[2];   // [L, H, 4H]
  const float* bih = (const float*)d_in[3];   // [L, 4H]
  const float* bhh = (const float*)d_in[4];   // [L, 4H]
  float* out = (float*)d_out;                 // [B*T*H] ++ h0,c0,h1,c1

  // ---- workspace carve-up ----
  char* ws = (char*)d_ws;
  size_t off = 0;
  auto carve = [&](size_t bytes) -> char* {
    char* p = ws + off;
    off = (off + bytes + 255) & ~(size_t)255;
    return p;
  };
  const size_t wbytes  = (size_t)LSTM_H * LSTM_G * sizeof(__bf16);          // 8 MB
  const size_t bhF     = (size_t)LSTM_B * LSTM_H * sizeof(float);           // 256 KB
  const size_t bhBF    = (size_t)LSTM_B * LSTM_H * sizeof(__bf16);          // 128 KB
  const size_t xbfB    = (size_t)LSTM_B * LSTM_T * LSTM_H * sizeof(__bf16); // 32 MB
  __bf16* wihP[2] = {(__bf16*)carve(wbytes), (__bf16*)carve(wbytes)};
  __bf16* whhP[2] = {(__bf16*)carve(wbytes), (__bf16*)carve(wbytes)};
  __bf16* xbf     = (__bf16*)carve(xbfB);
  float*  bsum    = (float*)carve((size_t)2 * LSTM_G * sizeof(float));
  __bf16* hbf[2][2];
  float*  hfull[2];
  float*  cbuf[2];
  for (int l = 0; l < 2; ++l) {
    hbf[l][0] = (__bf16*)carve(bhBF);
    hbf[l][1] = (__bf16*)carve(bhBF);
    hfull[l]  = (float*)carve(bhF);
    cbuf[l]   = (float*)carve(bhF);
  }

  // ---- one-time (per launch) conversions ----
  const int pgrid = (LSTM_H * LSTM_G) / 256;           // 16384 blocks
  for (int l = 0; l < 2; ++l) {
    const size_t woff = (size_t)l * LSTM_H * LSTM_G;
    pack_w_bf16<<<pgrid, 256, 0, stream>>>(Wih + woff, wihP[l]);
    pack_w_bf16<<<pgrid, 256, 0, stream>>>(Whh + woff, whhP[l]);
  }
  const long xn = (long)LSTM_B * LSTM_T * LSTM_H;      // 16,777,216
  to_bf16<<<(int)(xn / 256), 256, 0, stream>>>(x, xbf, xn);
  bias_sum<<<(2 * LSTM_G) / 256, 256, 0, stream>>>(bih, bhh, bsum, 2 * LSTM_G);

  // ---- zero initial states (bf16 zero == 0x0000) ----
  for (int l = 0; l < 2; ++l) {
    hipMemsetAsync(hbf[l][0], 0, bhBF, stream);
    hipMemsetAsync(cbuf[l],   0, bhF,  stream);
  }

  // ---- recurrence: 2 dependent launches per timestep ----
  const long xrs = (long)LSTM_T * LSTM_H;
  for (int t = 0; t < LSTM_T; ++t) {
    const int pr = t & 1, pw = pr ^ 1;
    // layer 0: input = xbf[:, t, :]
    lstm_step<<<32, 256, 0, stream>>>(
        xbf, xrs, (long)t * LSTM_H,
        hbf[0][pr], wihP[0], whhP[0], bsum,
        cbuf[0], hfull[0], hbf[0][pw],
        nullptr, 0L, 0L);
    // layer 1: input = layer-0 bf16 h just produced; writes output slice
    lstm_step<<<32, 256, 0, stream>>>(
        hbf[0][pw], (long)LSTM_H, 0L,
        hbf[1][pr], wihP[1], whhP[1], bsum + LSTM_G,
        cbuf[1], hfull[1], hbf[1][pw],
        out, xrs, (long)t * LSTM_H);
  }

  // ---- final states: outputs ++ [h_T0, c_T0, h_T1, c_T1] ----
  float* tail = out + (size_t)LSTM_B * LSTM_T * LSTM_H;
  const size_t bhN = (size_t)LSTM_B * LSTM_H;
  hipMemcpyAsync(tail + 0 * bhN, hfull[0], bhF, hipMemcpyDeviceToDevice, stream);
  hipMemcpyAsync(tail + 1 * bhN, cbuf[0],  bhF, hipMemcpyDeviceToDevice, stream);
  hipMemcpyAsync(tail + 2 * bhN, hfull[1], bhF, hipMemcpyDeviceToDevice, stream);
  hipMemcpyAsync(tail + 3 * bhN, cbuf[1],  bhF, hipMemcpyDeviceToDevice, stream);
}
